// NeuronEquivDeepSetLayer_11922829214367
// MI455X (gfx1250) — matrix operand order repaired and checked by reading, not compiled, not run on hardware
//
#include <hip/hip_runtime.h>

typedef __attribute__((ext_vector_type(16))) _Float16 v16h;
typedef __attribute__((ext_vector_type(8)))  _Float16 v8h;
typedef __attribute__((ext_vector_type(4)))  _Float16 v4h;
typedef __attribute__((ext_vector_type(8)))  float    v8f;
typedef __attribute__((ext_vector_type(4)))  float    v4f;

#define DDIM 128
#define NSEG 4096
#define MBLK 128
#define LDH  136   // padded LDS row stride in halves (272B -> 4-bank shift per row)
#define KT   4     // k-tiles of 32
#define NT   8     // n-tiles of 16

// ---------------------------------------------------------------------------
// Pack a 128x128 row-major f32 weight (K x N) into per-lane WMMA B fragments:
// frag[(kt*8+nt)*32 + lane][j] = w[kt*32 + (lane>>4)*8 + j + (j>=8 ? 8 : 0)][nt*16 + (lane&15)]
// so a wave reads one fragment as a fully-coalesced v16h (32B/lane) load.
// ---------------------------------------------------------------------------
__global__ __launch_bounds__(256) void pack_w16(const float* __restrict__ w,
                                                _Float16* __restrict__ wp) {
  int t    = blockIdx.x * 256 + threadIdx.x;   // 0..1023
  int lane = t & 31;
  int tile = t >> 5;                           // kt*8 + nt
  int kt = tile >> 3, nt = tile & 7;
  int n  = nt * 16 + (lane & 15);
  int kb = kt * 32 + (lane >> 4) * 8;
  _Float16* dst = wp + (size_t)t * 16;
#pragma unroll
  for (int j = 0; j < 16; ++j) dst[j] = (_Float16)w[(kb + j + (j >= 8 ? 8 : 0)) * DDIM + n];
}

__global__ __launch_bounds__(256) void zero_f32(float* p, int n) {
  int i = blockIdx.x * 256 + threadIdx.x;
  if (i < n) p[i] = 0.f;
}

// ---------------------------------------------------------------------------
// Segment sum: idx is sorted, so accumulate per-column locally and flush with
// a global f32 atomic only when the segment id changes (few atomics / block).
// ---------------------------------------------------------------------------
__global__ __launch_bounds__(128) void segsum128(const float* __restrict__ x,
                                                 const int* __restrict__ idx,
                                                 float* __restrict__ xsum,
                                                 int nrows) {
  const int ROWS = 1024;
  int col = threadIdx.x;                       // 0..127, one column per thread
  long long r0 = (long long)blockIdx.x * ROWS;
  long long rend = r0 + ROWS;
  if (rend > nrows) rend = nrows;
  if (r0 >= rend) return;
  float acc = 0.f;
  int cur = idx[r0];
  for (long long r = r0; r < rend; ++r) {
    int s = idx[r];                            // uniform across the block
    if (s != cur) {
      atomicAdd(&xsum[(size_t)cur * DDIM + col], acc);
      acc = 0.f;
      cur = s;
    }
    acc += x[(size_t)r * DDIM + col];
  }
  atomicAdd(&xsum[(size_t)cur * DDIM + col], acc);
}

// ---------------------------------------------------------------------------
// Fused 2-layer MLP (128->128->128) via v_wmma_f32_16x16x32_f16.
// Block = 256 threads = 8 waves; block tile = 128 rows; wave tile = 16 rows.
// Optional fused epilogue: out[row] = mlp(x[row]) + gsrc[gidx[row]].
// ---------------------------------------------------------------------------
__global__ __launch_bounds__(256)
void mlp128_wmma(const float* __restrict__ xin, int nrows,
                 const _Float16* __restrict__ w1p, const float* __restrict__ b1,
                 const _Float16* __restrict__ w2p, const float* __restrict__ b2,
                 const int* __restrict__ gidx, const float* __restrict__ gsrc,
                 float* __restrict__ out) {
  __shared__ _Float16 lbuf[MBLK * LDH];        // 34,816 B

  const int tid  = threadIdx.x;
  const int lane = tid & 31;
  const int wave = tid >> 5;                   // 0..7
  const size_t rowBase = (size_t)blockIdx.x * MBLK;

  // ---- stage x tile: f32 global -> f16 LDS (coalesced float4 loads) ----
  for (int i = tid; i < MBLK * (DDIM / 4); i += 256) {
    int r  = i >> 5;                           // row in tile
    int c4 = i & 31;                           // float4 index in row
    size_t gr = rowBase + (size_t)r;
    v4f v = {0.f, 0.f, 0.f, 0.f};
    if (gr < (size_t)nrows) v = *(const v4f*)(xin + gr * DDIM + c4 * 4);
    v4h h;
    h.x = (_Float16)v.x; h.y = (_Float16)v.y;
    h.z = (_Float16)v.z; h.w = (_Float16)v.w;
    *(v4h*)(&lbuf[r * LDH + c4 * 4]) = h;      // ds_store_b64
  }
  __syncthreads();

  const int m0   = wave * 16;                  // wave's row stripe
  const int arow = m0 + (lane & 15);           // A-frag: M = lane&15
  const int akb  = (lane >> 4) * 8;            // A-frag: K base per lane-half
  const int nlo  = lane & 15;                  // C/D: N = lane&15
  const int rhi  = (lane >> 4) * 8;            // C/D: M = v + 8*(lane>>4)

  const v16h* wv1 = (const v16h*)w1p;
  const v16h* wv2 = (const v16h*)w2p;

  // ---- layer 1: h = relu(x @ w1 + b1) ----
  v8f acc[NT];
#pragma unroll
  for (int nt = 0; nt < NT; ++nt) {
    float b = b1[nt * 16 + nlo];
#pragma unroll
    for (int j = 0; j < 8; ++j) acc[nt][j] = b;
  }
#pragma unroll
  for (int kt = 0; kt < KT; ++kt) {
    const _Float16* ap = &lbuf[arow * LDH + kt * 32 + akb];
    v8h a0 = *(const v8h*)ap;                  // K = kb+0..7   (ds_load_b128)
    v8h a1 = *(const v8h*)(ap + 16);           // K = kb+16..23 (ds_load_b128)
    v16h a = __builtin_shufflevector(a0, a1, 0,1,2,3,4,5,6,7,8,9,10,11,12,13,14,15);
#pragma unroll
    for (int nt = 0; nt < NT; ++nt) {
      v16h bf = wv1[(size_t)(kt * 8 + nt) * 32 + lane];
      acc[nt] = __builtin_amdgcn_wmma_f32_16x16x32_f16(false, a, false, bf,
                                                       (short)0, acc[nt], false, false);
    }
  }
  // ReLU -> f16 back into this wave's own LDS rows (wave-local reuse; LDS ops
  // from one wave stay in order, no cross-wave sharing of these rows).
#pragma unroll
  for (int nt = 0; nt < NT; ++nt) {
#pragma unroll
    for (int v = 0; v < 8; ++v) {
      float h = acc[nt][v];
      h = h > 0.f ? h : 0.f;
      lbuf[(m0 + v + rhi) * LDH + nt * 16 + nlo] = (_Float16)h;
    }
  }

  // ---- layer 2: y = h @ w2 + b2 ----
  v8f acc2[NT];
#pragma unroll
  for (int nt = 0; nt < NT; ++nt) {
    float b = b2[nt * 16 + nlo];
#pragma unroll
    for (int j = 0; j < 8; ++j) acc2[nt][j] = b;
  }
#pragma unroll
  for (int kt = 0; kt < KT; ++kt) {
    const _Float16* ap = &lbuf[arow * LDH + kt * 32 + akb];
    v8h a0 = *(const v8h*)ap;
    v8h a1 = *(const v8h*)(ap + 16);
    v16h a = __builtin_shufflevector(a0, a1, 0,1,2,3,4,5,6,7,8,9,10,11,12,13,14,15);
#pragma unroll
    for (int nt = 0; nt < NT; ++nt) {
      v16h bf = wv2[(size_t)(kt * 8 + nt) * 32 + lane];
      acc2[nt] = __builtin_amdgcn_wmma_f32_16x16x32_f16(false, a, false, bf,
                                                        (short)0, acc2[nt], false, false);
    }
  }

  // ---- epilogue: optional gather-add of rho_out[idx[row]], store f32 ----
#pragma unroll
  for (int v = 0; v < 8; ++v) {
    size_t grow = rowBase + (size_t)(m0 + v + rhi);
    if (grow < (size_t)nrows) {
      const float* gp = gidx ? (gsrc + (size_t)gidx[grow] * DDIM) : nullptr;
      float* op = out + grow * DDIM;
#pragma unroll
      for (int nt = 0; nt < NT; ++nt) {
        float val = acc2[nt][v];
        if (gp) val += gp[nt * 16 + nlo];
        op[nt * 16 + nlo] = val;
      }
    }
  }
}

// ---------------------------------------------------------------------------
extern "C" void kernel_launch(void* const* d_in, const int* in_sizes, int n_in,
                              void* d_out, int out_size, void* d_ws, size_t ws_size,
                              hipStream_t stream) {
  (void)n_in; (void)out_size; (void)ws_size;
  const float* x      = (const float*)d_in[0];
  const int*   idx    = (const int*)  d_in[1];
  const float* phi_w1 = (const float*)d_in[2];
  const float* phi_b1 = (const float*)d_in[3];
  const float* phi_w2 = (const float*)d_in[4];
  const float* phi_b2 = (const float*)d_in[5];
  const float* rho_w1 = (const float*)d_in[6];
  const float* rho_b1 = (const float*)d_in[7];
  const float* rho_w2 = (const float*)d_in[8];
  const float* rho_b2 = (const float*)d_in[9];
  float* out = (float*)d_out;

  const int nrows = in_sizes[0] / DDIM;        // 1,000,000

  char* ws = (char*)d_ws;
  float*    xsum    = (float*)ws;                                        // 2 MB
  float*    rho_out = (float*)(ws + (size_t)NSEG * DDIM * sizeof(float)); // 2 MB
  _Float16* pw1 = (_Float16*)(ws + 2 * (size_t)NSEG * DDIM * sizeof(float));
  _Float16* pw2 = pw1 + DDIM * DDIM;
  _Float16* pr1 = pw2 + DDIM * DDIM;
  _Float16* pr2 = pr1 + DDIM * DDIM;           // + 4 x 32 KB packed weights

  // pack all four weight matrices into WMMA B-fragment layout (f16)
  pack_w16<<<4, 256, 0, stream>>>(phi_w1, pw1);
  pack_w16<<<4, 256, 0, stream>>>(phi_w2, pw2);
  pack_w16<<<4, 256, 0, stream>>>(rho_w1, pr1);
  pack_w16<<<4, 256, 0, stream>>>(rho_w2, pr2);

  // segment sum (zero-init + sorted-run accumulation with boundary atomics)
  zero_f32<<<(NSEG * DDIM + 255) / 256, 256, 0, stream>>>(xsum, NSEG * DDIM);
  segsum128<<<(nrows + 1023) / 1024, 128, 0, stream>>>(x, idx, xsum, nrows);

  // rho MLP over the 4096 segment sums
  mlp128_wmma<<<NSEG / MBLK, 256, 0, stream>>>(xsum, NSEG, pr1, rho_b1, pr2, rho_b2,
                                               nullptr, nullptr, rho_out);

  // phi MLP over all nodes, fused with gather-add of rho_out[idx]
  mlp128_wmma<<<(nrows + MBLK - 1) / MBLK, 256, 0, stream>>>(
      x, nrows, pw1, phi_b1, pw2, phi_b2, idx, rho_out, out);
}
